// Sea_Attention_17626545783713
// MI455X (gfx1250) — matrix operand order repaired and checked by reading, not compiled, not run on hardware
//
#include <hip/hip_runtime.h>
#include <hip/hip_bf16.h>

// ---------------------------------------------------------------------------
// Types for CDNA5 WMMA (wave32, v_wmma_f32_16x16x32_bf16)
// ---------------------------------------------------------------------------
typedef __attribute__((ext_vector_type(16))) __bf16 v16bf;
typedef __attribute__((ext_vector_type(8)))  float  v8f;

union AF { v16bf v; unsigned int u[8]; uint4 q[2]; };

__device__ __forceinline__ unsigned short f2bf(float f) {
    unsigned int u = __float_as_uint(f);
    u = (u + 0x7FFFu + ((u >> 16) & 1u)) >> 16;   // round-to-nearest-even
    return (unsigned short)u;
}
__device__ __forceinline__ float bf2f(unsigned int h) {
    return __uint_as_float(h << 16);
}
// pack two fp32 -> bf16 pair, round-to-nearest (ties away): 2x v_add + v_perm
__device__ __forceinline__ unsigned int packrn(float a, float b) {
    return __builtin_amdgcn_perm(__float_as_uint(b) + 0x8000u,
                                 __float_as_uint(a) + 0x8000u, 0x07060302u);
}

// ---------------------------------------------------------------------------
// Workspace layout (bytes)
// ---------------------------------------------------------------------------
#define OFF_WPACK  ((size_t)0x0000000)  // bf16 packed weights, 1,114,112 ele
#define OFF_SB     ((size_t)0x0240000)  // fp32 packed scale/bias, 3072 ele
#define OFF_XMR    ((size_t)0x0280000)  // fp32 [16,256,64]
#define OFF_XMC    ((size_t)0x0380000)  // fp32 [16,256,64]
#define OFF_QKVR   ((size_t)0x0480000)  // fp32 [16,768,64]
#define OFF_QKVC   ((size_t)0x0780000)  // fp32 [16,768,64]
#define OFF_XXROW  ((size_t)0x0A80000)  // fp32 [16,512,64]
#define OFF_XXCOL  ((size_t)0x0C80000)  // fp32 [16,512,64]
#define OFF_XR     ((size_t)0x0E80000)  // fp32 [16,512,64]
#define OFF_XC     ((size_t)0x1080000)  // fp32 [16,512,64]
#define OFF_V      ((size_t)0x1280000)  // bf16 [16,512,4096]  (64 MiB)
#define OFF_G1     ((size_t)0x5280000)  // bf16 [16,256,4096]  (32 MiB)
#define OFF_DW     ((size_t)0x7280000)  // bf16 [16,256,4096]  (32 MiB)
#define WS_NEED    ((size_t)0x9280000)

// Packed-weight element offsets
#define WP_A0    0         // [wv;w_sc]  768x256
#define WP_A1    196608    // [wq;wk;wv] 768x256
#define WP_ROW   393216    // w_row 512x512
#define WP_COL   655360    // w_col 512x512
#define WP_PROJ  917504    // w_proj 256x512
#define WP_PW    1048576   // w_pw 256x256
#define WP_TOTAL 1114112

// ---------------------------------------------------------------------------
// Pack weights fp32 -> bf16 and scale/bias concatenations (one-time, RNE)
// ---------------------------------------------------------------------------
__global__ void pack_k(const float* __restrict__ wq, const float* __restrict__ wk,
                       const float* __restrict__ wv, const float* __restrict__ wsc,
                       const float* __restrict__ wrow, const float* __restrict__ wcol,
                       const float* __restrict__ wproj, const float* __restrict__ wpw,
                       const float* __restrict__ sq, const float* __restrict__ bq,
                       const float* __restrict__ sk, const float* __restrict__ bk,
                       const float* __restrict__ sv, const float* __restrict__ bv,
                       const float* __restrict__ ssc, const float* __restrict__ bsc,
                       unsigned short* __restrict__ wp, float* __restrict__ sb)
{
    int i = blockIdx.x * blockDim.x + threadIdx.x;
    if (i < WP_TOTAL) {
        float val;
        if (i < WP_A1) {                              // A0 = [wv ; w_sc]
            int r = i >> 8, k = i & 255;
            val = (r < 512) ? wv[r * 256 + k] : wsc[(r - 512) * 256 + k];
        } else if (i < WP_ROW) {                      // A1 = [wq ; wk ; wv]
            int j = i - WP_A1; int r = j >> 8, k = j & 255;
            val = (r < 128) ? wq[r * 256 + k]
                : (r < 256) ? wk[(r - 128) * 256 + k]
                            : wv[(r - 256) * 256 + k];
        } else if (i < WP_COL)  val = wrow[i - WP_ROW];
        else if (i < WP_PROJ)   val = wcol[i - WP_COL];
        else if (i < WP_PW)     val = wproj[i - WP_PROJ];
        else                    val = wpw[i - WP_PW];
        wp[i] = f2bf(val);
    }
    if (i < 3072) {
        int r = i & 767;
        float v;
        if (i < 768)        v = (r < 512) ? sv[r] : ssc[r - 512];          // s0
        else if (i < 1536)  v = (r < 512) ? bv[r] : bsc[r - 512];          // b0
        else if (i < 2304)  v = (r < 128) ? sq[r] : (r < 256 ? sk[r-128] : sv[r-256]); // s1
        else                v = (r < 128) ? bq[r] : (r < 256 ? bk[r-128] : bv[r-256]); // b1
        sb[i] = v;
    }
}

// ---------------------------------------------------------------------------
// Row / column means of x : one block per (b,c) plane of 64x64
// ---------------------------------------------------------------------------
__global__ void means_k(const float* __restrict__ x,
                        float* __restrict__ xmr, float* __restrict__ xmc)
{
    int bc = blockIdx.x;                   // 0..4095
    int t  = threadIdx.x;                  // 0..63
    const float* p = x + (size_t)bc * 4096;
    float cacc = 0.f;
    #pragma unroll 4
    for (int h = 0; h < 64; ++h) cacc += p[h * 64 + t];   // coalesced
    xmc[bc * 64 + t] = cacc * (1.f / 64.f);
    float racc = 0.f;
    #pragma unroll 4
    for (int w = 0; w < 64; ++w) racc += p[t * 64 + w];   // L2-resident re-read
    xmr[bc * 64 + t] = racc * (1.f / 64.f);
}

// ---------------------------------------------------------------------------
// Tiny axial attention: 256 problems (2 axes x 16 batch x 8 heads), n=64
// ---------------------------------------------------------------------------
__global__ void attn_k(const float* __restrict__ qkvr, const float* __restrict__ qkvc,
                       const float* __restrict__ prq, const float* __restrict__ prk,
                       const float* __restrict__ pcq, const float* __restrict__ pck,
                       float* __restrict__ xxrow, float* __restrict__ xxcol)
{
    int blk  = blockIdx.x;
    int axis = blk >> 7;
    int b    = (blk & 127) >> 3;
    int hd   = blk & 7;
    const float* qkv = axis ? qkvc : qkvr;
    const float* pq  = axis ? pcq  : prq;
    const float* pk  = axis ? pck  : prk;
    float* xx        = axis ? xxcol : xxrow;

    __shared__ float kS[16 * 64];
    __shared__ float vS[64 * 64];
    __shared__ float pS[64 * 65];          // padded: conflict-free per-row access

    int i = threadIdx.x;                   // position index 0..63
    // linear interpolation of pos embedding 16 -> 64 (align_corners=False)
    float coord = 0.25f * (float)i - 0.375f;
    coord = fminf(fmaxf(coord, 0.f), 15.f);
    int lo = (int)coord;
    int hi = (lo + 1 < 15) ? lo + 1 : 15;
    float t = coord - (float)lo;

    const float* kbase = qkv + ((size_t)b * 768 + 128 + hd * 16) * 64;
    #pragma unroll
    for (int c = 0; c < 16; ++c) {
        int ch = hd * 16 + c;
        float pv = pk[ch * 16 + lo] * (1.f - t) + pk[ch * 16 + hi] * t;
        kS[c * 64 + i] = kbase[c * 64 + i] + pv;
    }
    const float* vbase = qkv + ((size_t)b * 768 + 256 + hd * 64) * 64;
    for (int d = 0; d < 64; ++d) vS[d * 64 + i] = vbase[d * 64 + i];
    __syncthreads();

    float q[16];
    const float* qbase = qkv + ((size_t)b * 768 + hd * 16) * 64;
    #pragma unroll
    for (int c = 0; c < 16; ++c) {
        int ch = hd * 16 + c;
        float pv = pq[ch * 16 + lo] * (1.f - t) + pq[ch * 16 + hi] * t;
        q[c] = qbase[c * 64 + i] + pv;
    }
    float mx = -1e30f;
    for (int m = 0; m < 64; ++m) {
        float dot = 0.f;
        #pragma unroll
        for (int c = 0; c < 16; ++c) dot += q[c] * kS[c * 64 + m];
        dot *= 0.25f;                      // KEY_DIM^-0.5
        pS[i * 65 + m] = dot;
        mx = fmaxf(mx, dot);
    }
    float sum = 0.f;
    for (int m = 0; m < 64; ++m) {
        float e = __expf(pS[i * 65 + m] - mx);
        pS[i * 65 + m] = e;
        sum += e;
    }
    float inv = 1.f / sum;
    for (int d = 0; d < 64; ++d) {
        float a = 0.f;
        #pragma unroll 8
        for (int m = 0; m < 64; ++m) a += pS[i * 65 + m] * vS[d * 64 + m];
        xx[((size_t)b * 512 + hd * 64 + d) * 64 + i] = a * inv;
    }
}

// ---------------------------------------------------------------------------
// Depthwise 3x3 'SAME' conv + affine + relu on g1 (bf16 in/out)
// ---------------------------------------------------------------------------
__global__ void dw_k(const unsigned short* __restrict__ g1, const float* __restrict__ wdw,
                     const float* __restrict__ sdw, const float* __restrict__ bdw,
                     unsigned short* __restrict__ out)
{
    int bc = blockIdx.x;                    // b*256+c
    int c  = bc & 255;
    const unsigned short* p = g1 + (size_t)bc * 4096;
    unsigned short* o       = out + (size_t)bc * 4096;
    float w[9];
    #pragma unroll
    for (int j = 0; j < 9; ++j) w[j] = wdw[c * 9 + j];
    float s = sdw[c], bb = bdw[c];
    for (int pix = threadIdx.x; pix < 4096; pix += blockDim.x) {
        int h = pix >> 6, ww = pix & 63;
        float acc = 0.f;
        #pragma unroll
        for (int dy = -1; dy <= 1; ++dy) {
            int hh = h + dy; if (hh < 0 || hh > 63) continue;
            #pragma unroll
            for (int dx = -1; dx <= 1; ++dx) {
                int wx = ww + dx; if (wx < 0 || wx > 63) continue;
                acc += bf2f(p[hh * 64 + wx]) * w[(dy + 1) * 3 + (dx + 1)];
            }
        }
        o[pix] = f2bf(fmaxf(acc * s + bb, 0.f));
    }
}

// ---------------------------------------------------------------------------
// Generic bf16 WMMA GEMM:  D[m,n] = scale[m]*(A[m,:]·B[:,n]) + bias[m]
// Block tile 128x128, 8 waves (wave32), wave tile 64x32 = 4x2 WMMA tiles.
// Double-buffered LDS; all LDS traffic is 128-bit:
//   lA: [m][k]  rows padded to 40 bf16 (20 dw, 16B-aligned rows)
//   lB: [n][k]  pair-packed columns, rows padded to 40 bf16
// MODE selects B-source loader and epilogue:
//   0: B=x fp32            -> v bf16 (m<512) + g1 bf16 (m>=512)
//   1: B=xmean_row fp32    -> qkvr fp32            2: col variant -> qkvc
//   3: B=relu(xx_row) fp32 -> xr fp32              4: col variant -> xc
//   5: B=relu(v+xr+xc)     -> d_out fp32 (proj)
//   6: B=dw bf16           -> d_out *= h_sigmoid(val)   (gate fusion, in place)
// ---------------------------------------------------------------------------
template<int MODE>
__global__ __launch_bounds__(256)
void gemm_k(const unsigned short* __restrict__ wA,
            const float* __restrict__ scale, const float* __restrict__ bias,
            const float* __restrict__ srcF, const unsigned short* __restrict__ srcH,
            const float* __restrict__ auxR, const float* __restrict__ auxC,
            float* __restrict__ outF,
            unsigned short* __restrict__ outH, unsigned short* __restrict__ outH2)
{
    constexpr int K    = (MODE == 3 || MODE == 4 || MODE == 5) ? 512 : 256;
    constexpr int BUFQ = 128 * 5;                 // uint4s per LDS buffer
    __shared__ __attribute__((aligned(16))) unsigned short lA[2 * 128 * 40];
    __shared__ __attribute__((aligned(16))) unsigned short lB[2 * 128 * 40];

    const int tid   = threadIdx.x;
    const int mBase = blockIdx.y * 128;
    const int nBase = blockIdx.x * 128;
    const int wave  = tid >> 5;
    const int mw    = wave & 1;               // 2 waves along M (64 rows each)
    const int nw    = wave >> 1;              // 4 waves along N (32 cols each)
    const int l     = tid & 31;
    const int lm    = l & 15;
    const int grp   = l >> 4;

    v8f acc[4][2];
    #pragma unroll
    for (int mi = 0; mi < 4; ++mi)
        #pragma unroll
        for (int ni = 0; ni < 2; ++ni)
            #pragma unroll
            for (int r = 0; r < 8; ++r) acc[mi][ni][r] = 0.f;

    // A loader: thread owns half a row (16 bf16, 16B-vector IO)
    const int arow  = tid >> 1;
    const int ahalf = tid & 1;
    const unsigned short* gA = wA + (size_t)(mBase + arow) * K + ahalf * 16;
    uint4* lAq = (uint4*)lA;
    uint4* lBq = (uint4*)lB;

    // B loader: thread owns a column pair (n, n+1) and 8 consecutive k's
    const int np2 = (tid & 63) << 1;          // even column within tile
    const int kq  = tid >> 6;                 // k-quarter: k in [kq*8, kq*8+8)
    const int nG  = nBase + np2;
    int bB, sPix = 0, iPix = 0, hPix = 0, wPix = 0;
    if constexpr (MODE == 0 || MODE == 5 || MODE == 6) {
        bB = nG >> 12; sPix = nG & 4095; hPix = sPix >> 6; wPix = sPix & 63;
    } else {
        bB = nG >> 6;  iPix = nG & 63;
    }
    (void)hPix; (void)wPix;

    // ---- staging lambdas (registers only) ----
    auto loadA = [&](int k0, uint4& a0, uint4& a1) {
        const uint4* pa = (const uint4*)(gA + k0);
        a0 = pa[0]; a1 = pa[1];
    };
    auto loadB = [&](int k0, uint4& c0, uint4& c1) {
        const int kb = k0 + kq * 8;
        if constexpr (MODE == 6) {            // bf16 passthrough: pure v_perm
            const unsigned short* p = srcH + ((size_t)(bB << 8) + kb) * 4096 + sPix;
            unsigned int u[8];
            #pragma unroll
            for (int j = 0; j < 8; ++j) u[j] = *(const unsigned int*)(p + (size_t)j * 4096);
            c0.x = __builtin_amdgcn_perm(u[1], u[0], 0x05040100u);
            c0.y = __builtin_amdgcn_perm(u[3], u[2], 0x05040100u);
            c0.z = __builtin_amdgcn_perm(u[5], u[4], 0x05040100u);
            c0.w = __builtin_amdgcn_perm(u[7], u[6], 0x05040100u);
            c1.x = __builtin_amdgcn_perm(u[1], u[0], 0x07060302u);
            c1.y = __builtin_amdgcn_perm(u[3], u[2], 0x07060302u);
            c1.z = __builtin_amdgcn_perm(u[5], u[4], 0x07060302u);
            c1.w = __builtin_amdgcn_perm(u[7], u[6], 0x07060302u);
            if (k0 + 32 < K) __builtin_prefetch(p + (size_t)32 * 4096, 0, 3);
            return;
        }
        float v0[8], v1[8];
        if constexpr (MODE == 0) {
            const float* p = srcF + ((size_t)(bB << 8) + kb) * 4096 + sPix;
            #pragma unroll
            for (int j = 0; j < 8; ++j) {
                float2 tv = *(const float2*)(p + (size_t)j * 4096);
                v0[j] = tv.x; v1[j] = tv.y;
            }
            if (k0 + 32 < K) __builtin_prefetch(p + (size_t)32 * 4096, 0, 3);
        } else if constexpr (MODE == 1 || MODE == 2) {
            const float* p = srcF + ((size_t)(bB << 8) + kb) * 64 + iPix;
            #pragma unroll
            for (int j = 0; j < 8; ++j) {
                float2 tv = *(const float2*)(p + j * 64);
                v0[j] = tv.x; v1[j] = tv.y;
            }
        } else if constexpr (MODE == 3 || MODE == 4) {
            const float* p = srcF + ((size_t)(bB << 9) + kb) * 64 + iPix;
            #pragma unroll
            for (int j = 0; j < 8; ++j) {
                float2 tv = *(const float2*)(p + j * 64);
                v0[j] = fmaxf(tv.x, 0.f); v1[j] = fmaxf(tv.y, 0.f);
            }
        } else {                              // MODE == 5
            size_t ch0 = ((size_t)bB << 9) + kb;
            #pragma unroll
            for (int j = 0; j < 8; ++j) {
                size_t ch = ch0 + j;
                unsigned int uv = *(const unsigned int*)(srcH + ch * 4096 + sPix);
                float xrv = auxR[ch * 64 + hPix];
                float2 xcv = *(const float2*)(auxC + ch * 64 + wPix);
                v0[j] = fmaxf(bf2f(uv & 0xFFFFu) + xrv + xcv.x, 0.f);
                v1[j] = fmaxf(bf2f(uv >> 16)     + xrv + xcv.y, 0.f);
            }
            if (k0 + 32 < K)
                __builtin_prefetch(srcH + (ch0 + 32) * 4096 + sPix, 0, 3);
        }
        c0.x = packrn(v0[0], v0[1]); c0.y = packrn(v0[2], v0[3]);
        c0.z = packrn(v0[4], v0[5]); c0.w = packrn(v0[6], v0[7]);
        c1.x = packrn(v1[0], v1[1]); c1.y = packrn(v1[2], v1[3]);
        c1.z = packrn(v1[4], v1[5]); c1.w = packrn(v1[6], v1[7]);
    };
    auto stage = [&](int buf, uint4 a0, uint4 a1, uint4 c0, uint4 c1) {
        const int bo = buf * BUFQ;
        lAq[bo + arow * 5 + ahalf * 2 + 0] = a0;
        lAq[bo + arow * 5 + ahalf * 2 + 1] = a1;
        lBq[bo + np2 * 5 + kq]             = c0;
        lBq[bo + (np2 + 1) * 5 + kq]       = c1;
    };

    // ---- prologue: stage tile 0 ----
    uint4 a0, a1, c0, c1;
    loadA(0, a0, a1);
    loadB(0, c0, c1);
    stage(0, a0, a1, c0, c1);

    int cur = 0;
    for (int k0 = 0; k0 < K; k0 += 32) {
        __syncthreads();
        const bool more = (k0 + 32 < K);
        if (more) { loadA(k0 + 32, a0, a1); loadB(k0 + 32, c0, c1); }

        // ---- fragments: 2 x ds_load_b128 each (ISA 7.12.2 16-bit layouts) ----
        AF afr[4], bfr[2];
        const uint4* cA = (const uint4*)lA + cur * BUFQ;
        const uint4* cB = (const uint4*)lB + cur * BUFQ;
        #pragma unroll
        for (int mi = 0; mi < 4; ++mi) {
            int m = mw * 64 + mi * 16 + lm;
            afr[mi].q[0] = cA[m * 5 + grp];         // K 0-7   (this lane group)
            afr[mi].q[1] = cA[m * 5 + 2 + grp];     // K 16-23 (this lane group)
        }
        #pragma unroll
        for (int ni = 0; ni < 2; ++ni) {
            int n = nw * 32 + ni * 16 + lm;
            bfr[ni].q[0] = cB[n * 5 + grp];         // K 0-7   (this lane group)
            bfr[ni].q[1] = cB[n * 5 + 2 + grp];     // K 16-23 (this lane group)
        }
        #pragma unroll
        for (int mi = 0; mi < 4; ++mi)
            #pragma unroll
            for (int ni = 0; ni < 2; ++ni)
                acc[mi][ni] = __builtin_amdgcn_wmma_f32_16x16x32_bf16(
                    false, afr[mi].v, false, bfr[ni].v,
                    (short)0, acc[mi][ni], false, false);

        if (more) stage(cur ^ 1, a0, a1, c0, c1);
        cur ^= 1;
    }

    // ---- epilogue: per-channel affine + mode-specific store ----
    #pragma unroll
    for (int mi = 0; mi < 4; ++mi) {
        #pragma unroll
        for (int ni = 0; ni < 2; ++ni) {
            int nloc = nBase + nw * 32 + ni * 16 + lm;
            #pragma unroll
            for (int r = 0; r < 8; ++r) {
                int mg = mBase + mw * 64 + mi * 16 + r + grp * 8;
                float val = acc[mi][ni][r] * scale[mg] + bias[mg];
                if constexpr (MODE == 0) {
                    int b = nloc >> 12, s = nloc & 4095;
                    if (mg < 512) outH [(((size_t)b << 9) + mg) * 4096 + s] = f2bf(val);
                    else          outH2[(((size_t)b << 8) + (mg - 512)) * 4096 + s] = f2bf(val);
                } else if constexpr (MODE == 1 || MODE == 2) {
                    int b = nloc >> 6, i = nloc & 63;
                    outF[((size_t)b * 768 + mg) * 64 + i] = val;
                } else if constexpr (MODE == 3 || MODE == 4) {
                    int b = nloc >> 6, i = nloc & 63;
                    outF[(((size_t)b << 9) + mg) * 64 + i] = val;
                } else if constexpr (MODE == 5) {
                    int b = nloc >> 12, s = nloc & 4095;
                    outF[(((size_t)b << 8) + mg) * 4096 + s] = val;
                } else { // MODE == 6: h_sigmoid gate * attn_out (in place)
                    int b = nloc >> 12, s = nloc & 4095;
                    size_t idx = (((size_t)b << 8) + mg) * 4096 + s;
                    float g = fminf(fmaxf(val + 3.f, 0.f), 6.f) * (1.f / 6.f);
                    outF[idx] = g * outF[idx];
                }
            }
        }
    }
}

// ---------------------------------------------------------------------------
extern "C" void kernel_launch(void* const* d_in, const int* in_sizes, int n_in,
                              void* d_out, int out_size, void* d_ws, size_t ws_size,
                              hipStream_t stream)
{
    (void)in_sizes; (void)n_in; (void)out_size;
    if (ws_size < WS_NEED) return;

    const float* x     = (const float*)d_in[0];
    const float* wq    = (const float*)d_in[1];
    const float* sq    = (const float*)d_in[2];
    const float* bq    = (const float*)d_in[3];
    const float* wk    = (const float*)d_in[4];
    const float* sk    = (const float*)d_in[5];
    const float* bk    = (const float*)d_in[6];
    const float* wv    = (const float*)d_in[7];
    const float* sv    = (const float*)d_in[8];
    const float* bv    = (const float*)d_in[9];
    const float* prq   = (const float*)d_in[10];
    const float* prk   = (const float*)d_in[11];
    const float* pcq   = (const float*)d_in[12];
    const float* pck   = (const float*)d_in[13];
    const float* wrow  = (const float*)d_in[14];
    const float* srow  = (const float*)d_in[15];
    const float* brow  = (const float*)d_in[16];
    const float* wcol  = (const float*)d_in[17];
    const float* scol  = (const float*)d_in[18];
    const float* bcol  = (const float*)d_in[19];
    const float* wproj = (const float*)d_in[20];
    const float* sproj = (const float*)d_in[21];
    const float* bproj = (const float*)d_in[22];
    const float* wsc   = (const float*)d_in[23];
    const float* ssc   = (const float*)d_in[24];
    const float* bsc   = (const float*)d_in[25];
    const float* wdw   = (const float*)d_in[26];
    const float* sdw   = (const float*)d_in[27];
    const float* bdw   = (const float*)d_in[28];
    const float* wpw   = (const float*)d_in[29];
    const float* spw   = (const float*)d_in[30];
    const float* bpw   = (const float*)d_in[31];

    char* ws = (char*)d_ws;
    unsigned short* wp    = (unsigned short*)(ws + OFF_WPACK);
    float*          sb    = (float*)(ws + OFF_SB);
    float*          xmr   = (float*)(ws + OFF_XMR);
    float*          xmc   = (float*)(ws + OFF_XMC);
    float*          qkvr  = (float*)(ws + OFF_QKVR);
    float*          qkvc  = (float*)(ws + OFF_QKVC);
    float*          xxrow = (float*)(ws + OFF_XXROW);
    float*          xxcol = (float*)(ws + OFF_XXCOL);
    float*          xr    = (float*)(ws + OFF_XR);
    float*          xc    = (float*)(ws + OFF_XC);
    unsigned short* vbuf  = (unsigned short*)(ws + OFF_V);
    unsigned short* g1    = (unsigned short*)(ws + OFF_G1);
    unsigned short* dwb   = (unsigned short*)(ws + OFF_DW);
    float*          out   = (float*)d_out;

    // 1. pack weights + scale/bias
    pack_k<<<4352, 256, 0, stream>>>(wq, wk, wv, wsc, wrow, wcol, wproj, wpw,
                                     sq, bq, sk, bk, sv, bv, ssc, bsc, wp, sb);
    // 2. row/col means of x
    means_k<<<4096, 64, 0, stream>>>(x, xmr, xmc);
    // 3. v = cbn(x,wv), g1 = cbn(x,w_sc)   (fused 768-ch WMMA GEMM over 65536 px)
    gemm_k<0><<<dim3(512, 6), 256, 0, stream>>>(wp + WP_A0, sb, sb + 768,
                                                x, nullptr, nullptr, nullptr,
                                                nullptr, vbuf, g1);
    // 4. q/k/v means via GEMM on xmean (row & col)
    gemm_k<1><<<dim3(8, 6), 256, 0, stream>>>(wp + WP_A1, sb + 1536, sb + 2304,
                                              xmr, nullptr, nullptr, nullptr,
                                              qkvr, nullptr, nullptr);
    gemm_k<2><<<dim3(8, 6), 256, 0, stream>>>(wp + WP_A1, sb + 1536, sb + 2304,
                                              xmc, nullptr, nullptr, nullptr,
                                              qkvc, nullptr, nullptr);
    // 5. 256 tiny axial attentions -> xx_row / xx_col
    attn_k<<<256, 64, 0, stream>>>(qkvr, qkvc, prq, prk, pcq, pck, xxrow, xxcol);
    // 6. xr = cbn(relu(xx_row), w_row); xc = cbn(relu(xx_col), w_col)
    gemm_k<3><<<dim3(8, 4), 256, 0, stream>>>(wp + WP_ROW, srow, brow,
                                              xxrow, nullptr, nullptr, nullptr,
                                              xr, nullptr, nullptr);
    gemm_k<4><<<dim3(8, 4), 256, 0, stream>>>(wp + WP_COL, scol, bcol,
                                              xxcol, nullptr, nullptr, nullptr,
                                              xc, nullptr, nullptr);
    // 7. proj: d_out = cbn(relu(v + xr + xc), w_proj)   (B built on the fly)
    gemm_k<5><<<dim3(512, 2), 256, 0, stream>>>(wp + WP_PROJ, sproj, bproj,
                                                nullptr, vbuf, xr, xc,
                                                out, nullptr, nullptr);
    // 8. depthwise 3x3 + affine + relu on g1
    dw_k<<<4096, 256, 0, stream>>>(g1, wdw, sdw, bdw, dwb);
    // 9. pw GEMM + h_sigmoid gate, multiplied into d_out in place
    gemm_k<6><<<dim3(512, 2), 256, 0, stream>>>(wp + WP_PW, spw, bpw,
                                                nullptr, dwb, nullptr, nullptr,
                                                out, nullptr, nullptr);
}
